// PAdicAttention_23295902613806
// MI455X (gfx1250) — matrix operand order repaired and checked by reading, not compiled, not run on hardware
//
#include <hip/hip_runtime.h>
#include <hip/hip_bf16.h>

// ---------------------------------------------------------------------------
// PAdicAttention on MI455X (gfx1250), compile-only target.
//
// sim(q,k) = (1/5) * <F(q), F(k)> where F = concat_j tensor-prod_{i<=j}(p_i,1-p_i)
// (62 dims, padded to 64).  Heavy math runs through v_wmma_f32_16x16x32_bf16;
// attention K/V tiles are staged LDS-side by the Tensor Data Mover
// (tensor_load_to_lds + s_wait_tensorcnt) when the builtin is available.
// ---------------------------------------------------------------------------

typedef __bf16 bf16;
typedef __attribute__((ext_vector_type(16))) __bf16 v16bf;
typedef __attribute__((ext_vector_type(8)))  __bf16 bf16x8;
typedef __attribute__((ext_vector_type(8)))  float  v8f;
typedef unsigned int u32;
typedef __attribute__((ext_vector_type(4))) unsigned int u32x4;
typedef __attribute__((ext_vector_type(8))) int i32x8;
typedef __attribute__((ext_vector_type(4))) int i32x4;

constexpr int BATCH = 2;
constexpr int T     = 1024;
constexpr int DMODEL= 1024;
constexpr int H     = 8;
constexpr int DEPTH = 5;
constexpr int DH    = 128;   // d_v per head
constexpr int FD    = 64;    // padded feature dim (62 real + 2 zero)

#if __has_builtin(__builtin_amdgcn_tensor_load_to_lds) && \
    __has_builtin(__builtin_amdgcn_s_wait_tensorcnt)
#define PADIC_HAVE_TDM 1
#endif

// ---------------------------------------------------------------------------
// WMMA helpers.  Layouts per CDNA5 ISA §7.12.2 (wave32):
//  A (16x32 bf16): lane<16 -> row=lane, K {0..7,16..23}; lane>=16 -> K {8..15,24..31}
//  B (32x16 bf16): lane%16 -> column N, lane<16 -> K {0..7,16..23}; lane>=16 -> K {8..15,24..31}
//  C (16x16 f32):  vgpr j holds row j (lanes 0-15) / row j+8 (lanes 16-31), col=lane%16
// ---------------------------------------------------------------------------

__device__ __forceinline__ v16bf make16(bf16x8 lo, bf16x8 hi) {
  v16bf r;
#pragma unroll
  for (int i = 0; i < 8; ++i) { r[i] = lo[i]; r[i + 8] = hi[i]; }
  return r;
}

__device__ __forceinline__ v16bf load_a_frag(const bf16* buf, int stride, int lane) {
  int r = lane & 15;
  int s = (lane >> 4) << 3;                  // 0 or 8
  const bf16* p = buf + r * stride + s;
  bf16x8 lo = *(const bf16x8*)p;             // K = s .. s+7
  bf16x8 hi = *(const bf16x8*)(p + 16);      // K = s+16 .. s+23
  return make16(lo, hi);
}

__device__ __forceinline__ v16bf load_b_frag(const bf16* bufT, int stride, int lane) {
  int n = lane & 15;
  int s = (lane >> 4) << 3;                  // 0 or 8
  const bf16* p = bufT + n * stride + s;
  bf16x8 lo = *(const bf16x8*)p;
  bf16x8 hi = *(const bf16x8*)(p + 16);
  return make16(lo, hi);
}

__device__ __forceinline__ v8f wmma_bf16(v16bf a, v16bf b, v8f c) {
  return __builtin_amdgcn_wmma_f32_16x16x32_bf16(false, a, false, b, (short)0, c,
                                                 false, false);
}

// ---------------------------------------------------------------------------
// Tensor Data Mover: 2D tile load, 16-bit elements, optional LDS row padding.
// Descriptor bitfields per CDNA5 ISA.  This
// toolchain exposes the 6-arg builtin (extra descriptor-group vector).
// ---------------------------------------------------------------------------
#ifdef PADIC_HAVE_TDM
__device__ __forceinline__ void tdm_load_2d(u32 lds_addr, const void* gptr,
                                            u32 tile_d0, u32 tile_d1,
                                            u32 tensor_d0, u32 tensor_d1,
                                            u32 stride0, u32 pad_interval,
                                            u32 pad_amount, u32 pad_en) {
  unsigned long long ga = (unsigned long long)(uintptr_t)gptr;
  u32x4 g0;
  g0.x = 1u;                                            // count=1, user mode
  g0.y = lds_addr;                                      // lds_addr[31:0]
  g0.z = (u32)ga;                                       // global_addr[31:0]
  g0.w = (u32)((ga >> 32) & 0x1FFFFFFu) | (2u << 30);   // addr[56:32] | type=2
  i32x8 g1;
  g1[0] = (int)((1u << 16) |                            // data_size = 2 bytes
                (pad_en << 20) | (pad_interval << 22) | (pad_amount << 25));
  g1[1] = (int)((tensor_d0 & 0xFFFFu) << 16);           // tensor_dim0 lo16
  g1[2] = (int)(((tensor_d0 >> 16) & 0xFFFFu) |         // tensor_dim0 hi16
                ((tensor_d1 & 0xFFFFu) << 16));         // tensor_dim1 lo16
  g1[3] = (int)(((tensor_d1 >> 16) & 0xFFFFu) |         // tensor_dim1 hi16
                ((tile_d0 & 0xFFFFu) << 16));           // tile_dim0
  g1[4] = (int)(tile_d1 & 0xFFFFu);                     // tile_dim1 (tile_dim2=0)
  g1[5] = (int)stride0;                                 // tensor_dim0_stride lo32
  g1[6] = 0;
  g1[7] = 0;
  i32x4 z4 = {0, 0, 0, 0};                              // groups 2/3 unused (2D)
  i32x8 z8 = {0, 0, 0, 0, 0, 0, 0, 0};                  // trailing group unused
  __builtin_amdgcn_tensor_load_to_lds(g0, g1, z4, z4, z8, 0);
}
#endif

// ---------------------------------------------------------------------------
// Kernel 1: paths = sigmoid(x@Wp + bp); expand to 62-dim features, scale by
// 1/sqrt(5), write bf16 F laid out (B, H, T, 64).  One block per (b,t) row.
// ---------------------------------------------------------------------------
__global__ __launch_bounds__(64)
void padic_paths_features(const float* __restrict__ x,
                          const float* __restrict__ Wp,
                          const float* __restrict__ bp,
                          bf16* __restrict__ F) {
  __shared__ float xrow[DMODEL];
  __shared__ float pv[H * DEPTH];
  const int row = blockIdx.x;                // b*T + t
  const int tid = threadIdx.x;
  const float* xr = x + (size_t)row * DMODEL;

  for (int i = tid; i < DMODEL / 4; i += 64)
    ((float4*)xrow)[i] = ((const float4*)xr)[i];
  __syncthreads();

  if (tid < H * DEPTH) {
    float acc = bp[tid];
    for (int k = 0; k < DMODEL; ++k)
      acc = fmaf(xrow[k], Wp[k * (H * DEPTH) + tid], acc);
    pv[tid] = 1.0f / (1.0f + __expf(-acc));
  }
  __syncthreads();

  if (tid < H) {
    const int b = row / T, t = row - b * T;
    bf16* out = F + (((size_t)(b * H + tid)) * T + t) * FD;
    const float sc = 0.44721359549995794f;   // 1/sqrt(DEPTH)
    float buf[32];
    buf[0] = 1.0f;
    int L = 1, off = 0;
#pragma unroll
    for (int d = 0; d < DEPTH; ++d) {
      float pd = pv[tid * DEPTH + d];
      for (int i = L - 1; i >= 0; --i) {
        float v = buf[i];
        buf[i]     = v * pd;
        buf[i + L] = v * (1.0f - pd);
      }
      L <<= 1;
      for (int i = 0; i < L; ++i) out[off + i] = (bf16)(buf[i] * sc);
      off += L;                              // 2+4+8+16+32 = 62
    }
    out[62] = (bf16)0.0f;
    out[63] = (bf16)0.0f;
  }
}

// ---------------------------------------------------------------------------
// Kernel 2/4: bf16 WMMA GEMM, C(2048x1024) = A(2048x1024) @ B(1024x1024).
// AMODE 0: A row-major f32.          AMODE 1: A gathered from ctx (B,H,T,DH).
// OMODE 0: f32 row-major out.        OMODE 1: bf16 scatter to key-tiled V
//                                             layout (B*H, T/32, DH, 32).
// Block 256 thr (8 waves); tile 128x128x32; wave tile 32x64 (8 wmma / k-step).
// ---------------------------------------------------------------------------
template <int AMODE, int OMODE>
__global__ __launch_bounds__(256)
void padic_gemm(const float* __restrict__ A, const float* __restrict__ Bm,
                void* __restrict__ Cout) {
  constexpr int N = DMODEL, K = DMODEL;
  constexpr int BM = 128, BN = 128, BK = 32;
  constexpr int LS = BK + 8;                 // 40 elems = 80B rows (16B aligned)
  __shared__ bf16 sA[BM][LS];                // [m][k]
  __shared__ bf16 sB[BN][LS];                // transposed [n][k]

  const int tid = threadIdx.x, lane = tid & 31, w = tid >> 5;
  const int m0 = blockIdx.y * BM, n0 = blockIdx.x * BN;
  const int wm = (w >> 1) * 32;              // 0/32/64/96
  const int wn = (w & 1) * 64;               // 0/64

  v8f acc[2][4];
#pragma unroll
  for (int mi = 0; mi < 2; ++mi)
#pragma unroll
    for (int ni = 0; ni < 4; ++ni) acc[mi][ni] = v8f{0, 0, 0, 0, 0, 0, 0, 0};

  for (int kk = 0; kk < K; kk += BK) {
    for (int idx = tid; idx < BM * BK; idx += 256) {
      int r = idx >> 5, c = idx & 31;
      int row = m0 + r, k = kk + c;
      float v;
      if (AMODE == 0) {
        v = A[(size_t)row * K + k];
      } else {                               // ctx (B,H,T,DH) gather
        int b = row >> 10, t = row & 1023, h = k >> 7, dh = k & 127;
        v = A[(((size_t)(b * H + h)) * T + t) * DH + dh];
      }
      sA[r][c] = (bf16)v;
    }
    for (int idx = tid; idx < BK * BN; idx += 256) {
      int k = idx >> 7, n = idx & 127;
      sB[n][k] = (bf16)Bm[(size_t)(kk + k) * N + n0 + n];
    }
    if (kk + BK < K)                          // global_prefetch_b8 path
      __builtin_prefetch(&Bm[(size_t)(kk + BK) * N + n0 + (tid & 127)], 0, 1);
    __syncthreads();

    v16bf af0 = load_a_frag(&sA[wm][0],      LS, lane);
    v16bf af1 = load_a_frag(&sA[wm + 16][0], LS, lane);
#pragma unroll
    for (int ni = 0; ni < 4; ++ni) {
      v16bf bfrag = load_b_frag(&sB[wn + ni * 16][0], LS, lane);
      acc[0][ni] = wmma_bf16(af0, bfrag, acc[0][ni]);
      acc[1][ni] = wmma_bf16(af1, bfrag, acc[1][ni]);
    }
    __syncthreads();
  }

  const int cn = lane & 15, hi = lane >> 4;
#pragma unroll
  for (int mi = 0; mi < 2; ++mi) {
#pragma unroll
    for (int ni = 0; ni < 4; ++ni) {
#pragma unroll
      for (int j = 0; j < 8; ++j) {
        int row = m0 + wm + mi * 16 + j + 8 * hi;
        int col = n0 + wn + ni * 16 + cn;
        float v = acc[mi][ni][j];
        if (OMODE == 0) {
          ((float*)Cout)[(size_t)row * N + col] = v;
        } else {                             // key-tiled V: (B*H, T/32, DH, 32)
          int b = row >> 10, t = row & 1023, h = col >> 7, dh = col & 127;
          size_t bh = (size_t)(b * H + h);
          ((bf16*)Cout)[((bh * (T / 32) + (t >> 5)) * DH + dh) * 32 + (t & 31)]
              = (bf16)v;
        }
      }
    }
  }
}

// ---------------------------------------------------------------------------
// Kernel 3: causal flash attention.  Q=K=F (T x 64 bf16, pre-scaled), V bf16
// key-tiled (B*H, T/32, DH, 32).  Block = 128 q rows (8 waves x 16), 32-key
// tiles.  K tile -> LDS [key][feat] and V tile -> LDS [dh][key] are contiguous
// copies done by the Tensor Data Mover (TDM adds the LDS bank padding); P
// converts C->A layout through per-wave LDS scratch.
// ---------------------------------------------------------------------------
__global__ __launch_bounds__(256)
void padic_flash_attn(const bf16* __restrict__ F, const bf16* __restrict__ V,
                      float* __restrict__ ctx) {
  constexpr int KS = 72;                     // 128B data + 16B pad per row
  constexpr int VS = 40;                     // 64B data + 16B pad per row
  constexpr int PS = 40;
  __shared__ bf16 sK[32][KS];                // [key][feature]
  __shared__ bf16 sVT[DH][VS];               // [dh][key]
  __shared__ bf16 sP[8][16][PS];             // per-wave P scratch [qrow][key]

  const int tid = threadIdx.x, lane = tid & 31, w = tid >> 5;
  const int bh = blockIdx.y;                 // b*H + h
  const int qbase = blockIdx.x * 128;
  const int q0 = qbase + w * 16;

  const bf16* Fh = F + (size_t)bh * T * FD;
  const bf16* Vh = V + (size_t)bh * (T / 32) * DH * 32;

  // Q fragments (held in registers for the whole k loop).
  v16bf qa0 = load_a_frag(Fh + (size_t)q0 * FD + 0,  FD, lane);
  v16bf qa1 = load_a_frag(Fh + (size_t)q0 * FD + 32, FD, lane);

  v8f o[8];
#pragma unroll
  for (int n = 0; n < 8; ++n) o[n] = v8f{0, 0, 0, 0, 0, 0, 0, 0};
  float m8[8], l8[8];
#pragma unroll
  for (int j = 0; j < 8; ++j) { m8[j] = -1.0e30f; l8[j] = 0.0f; }

  const int kend = qbase + 128;
  for (int kt = 0; kt < kend; kt += 32) {
    const bf16* Vt = Vh + (size_t)(kt >> 5) * DH * 32;
#ifdef PADIC_HAVE_TDM
    if (w == 0) {
      // K tile: 32 rows x 64 bf16 (128B) + 16B LDS pad -> interval enc 4 (32
      // DWORDs), amount enc 3 (4 DWORDs).
      tdm_load_2d((u32)(uintptr_t)&sK[0][0], Fh + (size_t)kt * FD,
                  /*tile*/ FD, 32, /*tensor*/ FD, (u32)T, /*stride0*/ FD,
                  /*pad_interval*/ 4, /*pad_amount*/ 3, 1);
      // V tile: 128 rows x 32 bf16 (64B) + 16B LDS pad -> enc 3 / enc 3.
      tdm_load_2d((u32)(uintptr_t)&sVT[0][0], Vt,
                  /*tile*/ 32, DH, /*tensor*/ 32, (u32)(DH * (T / 32)),
                  /*stride0*/ 32, /*pad_interval*/ 3, /*pad_amount*/ 3, 1);
      __builtin_amdgcn_s_wait_tensorcnt(0);
    }
#else
    {                                        // fallback: manual staging
      int r = tid >> 3, c8 = (tid & 7) * 8;
      *(bf16x8*)&sK[r][c8] = *(const bf16x8*)(Fh + (size_t)(kt + r) * FD + c8);
    }
    for (int idx = tid; idx < DH * 32; idx += 256) {
      int dh = idx >> 5, key = idx & 31;
      sVT[dh][key] = Vt[idx];
    }
#endif
    __syncthreads();

    if (kt <= q0 + 15) {                     // causal: this wave has live keys
      v8f s0 = v8f{0, 0, 0, 0, 0, 0, 0, 0};
      v8f s1 = v8f{0, 0, 0, 0, 0, 0, 0, 0};
      s0 = wmma_bf16(qa0, load_b_frag(&sK[0][0],   KS, lane), s0);
      s0 = wmma_bf16(qa1, load_b_frag(&sK[0][32],  KS, lane), s0);
      s1 = wmma_bf16(qa0, load_b_frag(&sK[16][0],  KS, lane), s1);
      s1 = wmma_bf16(qa1, load_b_frag(&sK[16][32], KS, lane), s1);

      const int cn = lane & 15, hi = lane >> 4;
      const int rowg = q0 + 8 * hi;
      float scj[8];
#pragma unroll
      for (int j = 0; j < 8; ++j) {
        int r = rowg + j;
        float a0 = (kt + cn      <= r) ? s0[j] : -1.0e30f;
        float a1 = (kt + 16 + cn <= r) ? s1[j] : -1.0e30f;
        float mx = fmaxf(a0, a1);
#pragma unroll
        for (int off = 1; off < 16; off <<= 1)
          mx = fmaxf(mx, __shfl_xor(mx, off, 16));
        float mn = fmaxf(m8[j], mx);
        float sc = __expf(m8[j] - mn);
        float p0 = __expf(a0 - mn);
        float p1 = __expf(a1 - mn);
        float rs = p0 + p1;
#pragma unroll
        for (int off = 1; off < 16; off <<= 1)
          rs += __shfl_xor(rs, off, 16);
        l8[j] = l8[j] * sc + rs;
        m8[j] = mn;
        scj[j] = sc;
        sP[w][j + 8 * hi][cn]      = (bf16)p0;   // C-layout -> LDS
        sP[w][j + 8 * hi][16 + cn] = (bf16)p1;
      }
#pragma unroll
      for (int n = 0; n < 8; ++n)
#pragma unroll
        for (int j = 0; j < 8; ++j) o[n][j] *= scj[j];

      __builtin_amdgcn_wave_barrier();       // keep P store/load ordered
      v16bf pa = load_a_frag(&sP[w][0][0], PS, lane);
#pragma unroll
      for (int n = 0; n < 8; ++n) {
        v16bf bv = load_b_frag(&sVT[n * 16][0], VS, lane);
        o[n] = wmma_bf16(pa, bv, o[n]);
      }
    }
    __syncthreads();
  }

  const int cn = lane & 15, hi = lane >> 4;
#pragma unroll
  for (int j = 0; j < 8; ++j) {
    float inv = 1.0f / l8[j];
    int row = q0 + j + 8 * hi;
    float* dst = ctx + ((size_t)bh * T + row) * DH;
#pragma unroll
    for (int n = 0; n < 8; ++n) dst[n * 16 + cn] = o[n][j] * inv;
  }
}

// ---------------------------------------------------------------------------
// Launch.  Workspace: F (2MB bf16) | Vb (4MB bf16) | ctx (8MB f32).
// ---------------------------------------------------------------------------
extern "C" void kernel_launch(void* const* d_in, const int* in_sizes, int n_in,
                              void* d_out, int out_size, void* d_ws,
                              size_t ws_size, hipStream_t stream) {
  const float* x  = (const float*)d_in[0];
  const float* Wp = (const float*)d_in[1];
  const float* bp = (const float*)d_in[2];
  const float* Wv = (const float*)d_in[3];
  const float* Wo = (const float*)d_in[4];
  float* out = (float*)d_out;

  char* ws = (char*)d_ws;
  bf16*  F   = (bf16*)ws;                          // B*H*T*64  bf16 = 2 MB
  bf16*  Vb  = (bf16*)(ws + (size_t)(2 << 20));    // key-tiled V, 4 MB
  float* ctx = (float*)(ws + (size_t)(6 << 20));   // B*H*T*128 f32  = 8 MB

  padic_paths_features<<<BATCH * T, 64, 0, stream>>>(x, Wp, bp, F);
  padic_gemm<0, 1><<<dim3(DMODEL / 128, (BATCH * T) / 128), 256, 0, stream>>>(
      x, Wv, (void*)Vb);
  padic_flash_attn<<<dim3(T / 128, BATCH * H), 256, 0, stream>>>(F, Vb, ctx);
  padic_gemm<1, 0><<<dim3(DMODEL / 128, (BATCH * T) / 128), 256, 0, stream>>>(
      ctx, Wo, (void*)out);
}
